// MACLOSiamNet_41051297415255
// MI455X (gfx1250) — matrix operand order repaired and checked by reading, not compiled, not run on hardware
//
#include <hip/hip_runtime.h>
#include <hip/hip_bf16.h>
#include <math.h>

// MI455X / gfx1250: wave32, WMMA 16x16x32 bf16 (f32 accum), 320KB LDS per WGP.

#define EPSV 1e-5f

typedef __attribute__((ext_vector_type(16))) __bf16 v16bf;
typedef __attribute__((ext_vector_type(8)))  float  v8f;

// ---- CDNA5 WMMA fragment index helpers (ISA 7.12.2, wave32) ----
// 16-bit A-matrix 16x32: lane<16 -> K in {0..7, 16..23}; lane>=16 -> K in {8..15, 24..31}; M = lane&15
__device__ __forceinline__ int a_kidx(int e, int lane) {
  int koff = (lane & 16) ? 8 : 0;
  return (e < 8) ? (e + koff) : (e + 8 + koff);
}
// 16-bit B-matrix 32x16: lane<16 -> K in {0..7, 16..23}; lane>=16 -> K in {8..15, 24..31}; N = lane&15
__device__ __forceinline__ int b_kidx(int e, int lane) {
  int koff = (lane & 16) ? 8 : 0;
  return (e < 8) ? (e + koff) : (e + 8 + koff);
}
// f32 C/D 16x16: VGPR r -> M = r + 8*(lane>=16); N = lane&15
__device__ __forceinline__ int cd_row(int r, int lane) {
  return r + ((lane & 16) ? 8 : 0);
}
__device__ __forceinline__ v8f wmma_bf16(v16bf a, v16bf b, v8f c) {
  return __builtin_amdgcn_wmma_f32_16x16x32_bf16(false, a, false, b, (short)0, c, false, false);
}
// Packed fragment: one 32B contiguous vector load per lane (coalesced 1KB/wave)
__device__ __forceinline__ v16bf load_frag(const __bf16* packed, int fragIdx, int lane) {
  return *(const v16bf*)(packed + ((fragIdx * 32 + lane) << 4));
}

// ---- Kernel 0: pre-swizzle the five 256x256 weights into B-fragment layout (bf16) ----
__global__ void __launch_bounds__(256)
pack_weights(const float* __restrict__ Wp, const float* __restrict__ Wq,
             const float* __restrict__ Wk, const float* __restrict__ Wv,
             const float* __restrict__ Wo, __bf16* __restrict__ dst) {
  int tid = blockIdx.x * 256 + threadIdx.x;           // 5 * 65536 total
  int mat = tid >> 16;
  int r   = tid & 65535;                              // ((nt*8+kt)*32+lane)*16+e
  int e = r & 15, lane = (r >> 4) & 31, kt = (r >> 9) & 7, nt = r >> 12;
  int k = kt * 32 + b_kidx(e, lane);
  int n = nt * 16 + (lane & 15);
  const float* W = (mat == 0) ? Wp : (mat == 1) ? Wq : (mat == 2) ? Wk : (mat == 3) ? Wv : Wo;
  dst[tid] = (__bf16)W[k * 256 + n];
}

// ---- Kernel 1: patch-embed GEMM (16x256x256 per WG) + LayerNorm + NeuroFocal scale ----
__global__ void __launch_bounds__(256)
patch_embed_nfnorm(const float* __restrict__ x, const __bf16* __restrict__ Wp_pk,
                   const float* __restrict__ bp, const float* __restrict__ ln_g,
                   const float* __restrict__ ln_b, float* __restrict__ z) {
  const int b = blockIdx.x >> 4;
  const int tile = blockIdx.x & 15;
  const int t = threadIdx.x;
  const int w = t >> 5, lane = t & 31;
  __shared__ float zt[16][256];
  __shared__ float ps1[16][16], ps2[16][16];
  __shared__ float mu[16], rstd[16], mu2[16], den[16], scl[16];

  v8f acc0 = {}, acc1 = {};
  const int tok = tile * 16 + (lane & 15);
  const int ph = tok >> 4, pw = tok & 15;
  const float* xb = x + b * 65536;
  for (int kt = 0; kt < 8; ++kt) {
    v16bf a;
#pragma unroll
    for (int e = 0; e < 16; ++e) {
      int kk = kt * 32 + a_kidx(e, lane);             // k = pr*16 + pc (C=1)
      int pr = kk >> 4, pc = kk & 15;
      a[e] = (__bf16)xb[(ph * 16 + pr) * 256 + pw * 16 + pc];
    }
    v16bf b0 = load_frag(Wp_pk, (w * 2 + 0) * 8 + kt, lane);
    v16bf b1 = load_frag(Wp_pk, (w * 2 + 1) * 8 + kt, lane);
    acc0 = wmma_bf16(a, b0, acc0);
    acc1 = wmma_bf16(a, b1, acc1);
  }
#pragma unroll
  for (int r = 0; r < 8; ++r) {
    int m = cd_row(r, lane);
    int n0 = w * 32 + (lane & 15), n1 = w * 32 + 16 + (lane & 15);
    zt[m][n0] = acc0[r] + bp[n0];
    zt[m][n1] = acc1[r] + bp[n1];
  }
  __syncthreads();

  const int tk = t >> 4, seg = t & 15;
  float s1 = 0.f, s2 = 0.f;
  for (int d = seg * 16; d < seg * 16 + 16; ++d) { float v = zt[tk][d]; s1 += v; s2 += v * v; }
  ps1[tk][seg] = s1; ps2[tk][seg] = s2;
  __syncthreads();
  if (t < 16) {
    float S1 = 0.f, S2 = 0.f;
    for (int s = 0; s < 16; ++s) { S1 += ps1[t][s]; S2 += ps2[t][s]; }
    float m_ = S1 * (1.f / 256.f);
    float v_ = S2 * (1.f / 256.f) - m_ * m_;
    mu[t] = m_; rstd[t] = rsqrtf(v_ + EPSV);
  }
  __syncthreads();
  s1 = 0.f; s2 = 0.f;
  for (int d = seg * 16; d < seg * 16 + 16; ++d) {
    float v = (zt[tk][d] - mu[tk]) * rstd[tk] * ln_g[d] + ln_b[d];
    zt[tk][d] = v; s1 += v; s2 += v * v;
  }
  ps1[tk][seg] = s1; ps2[tk][seg] = s2;
  __syncthreads();
  if (t < 16) {
    float S1 = 0.f, S2 = 0.f;
    for (int s = 0; s < 16; ++s) { S1 += ps1[t][s]; S2 += ps2[t][s]; }
    float m_ = S1 * (1.f / 256.f);
    float v_ = S2 * (1.f / 256.f) - m_ * m_;
    mu2[t] = m_; den[t] = sqrtf(v_ + EPSV) + EPSV;
  }
  __syncthreads();
  s1 = 0.f;
  for (int d = seg * 16; d < seg * 16 + 16; ++d) s1 += fabsf(zt[tk][d] - mu2[tk]);
  ps1[tk][seg] = s1;
  __syncthreads();
  if (t < 16) {
    float S = 0.f;
    for (int s = 0; s < 16; ++s) S += ps1[t][s];
    scl[t] = 1.f + S / den[t];
  }
  __syncthreads();
  float* zrow = z + ((b * 256 + tile * 16 + tk) * 256);
  for (int d = seg * 16; d < seg * 16 + 16; ++d) zrow[d] = zt[tk][d] * scl[tk];
}

// ---- Kernel 2: fused Q/K/V GEMMs (shared A-fragment, 3 B matrices) ----
__global__ void __launch_bounds__(256)
qkv_gemm(const float* __restrict__ z, const __bf16* __restrict__ Wq_pk,
         const __bf16* __restrict__ Wk_pk, const __bf16* __restrict__ Wv_pk,
         const float* __restrict__ bq, const float* __restrict__ bk,
         const float* __restrict__ bv, float* __restrict__ Q,
         float* __restrict__ K, float* __restrict__ V) {
  const int b = blockIdx.x >> 4;
  const int tile = blockIdx.x & 15;
  const int w = threadIdx.x >> 5, lane = threadIdx.x & 31;
  const int nt0 = w * 2, nt1 = w * 2 + 1;
  const float* zrow = z + ((b * 256 + tile * 16 + (lane & 15)) * 256);
  v8f aq0 = {}, aq1 = {}, ak0 = {}, ak1 = {}, av0 = {}, av1 = {};
  for (int kt = 0; kt < 8; ++kt) {
    v16bf a;
#pragma unroll
    for (int e = 0; e < 16; ++e) a[e] = (__bf16)zrow[kt * 32 + a_kidx(e, lane)];
    v16bf q0 = load_frag(Wq_pk, nt0 * 8 + kt, lane), q1 = load_frag(Wq_pk, nt1 * 8 + kt, lane);
    v16bf k0 = load_frag(Wk_pk, nt0 * 8 + kt, lane), k1 = load_frag(Wk_pk, nt1 * 8 + kt, lane);
    v16bf v0 = load_frag(Wv_pk, nt0 * 8 + kt, lane), v1 = load_frag(Wv_pk, nt1 * 8 + kt, lane);
    aq0 = wmma_bf16(a, q0, aq0); aq1 = wmma_bf16(a, q1, aq1);
    ak0 = wmma_bf16(a, k0, ak0); ak1 = wmma_bf16(a, k1, ak1);
    av0 = wmma_bf16(a, v0, av0); av1 = wmma_bf16(a, v1, av1);
  }
#pragma unroll
  for (int r = 0; r < 8; ++r) {
    int m = cd_row(r, lane);
    int row = (b * 256 + tile * 16 + m) * 256;
    int n0 = nt0 * 16 + (lane & 15), n1 = nt1 * 16 + (lane & 15);
    Q[row + n0] = aq0[r] + bq[n0];  Q[row + n1] = aq1[r] + bq[n1];
    K[row + n0] = ak0[r] + bk[n0];  K[row + n1] = ak1[r] + bk[n1];
    V[row + n0] = av0[r] + bv[n0];  V[row + n1] = av1[r] + bv[n1];
  }
}

// ---- Kernel 3: per-(b,h,n) row stats: |Q|^2, |K|^2, omega multiplier 1/(e^{-nf_q}+1) ----
__global__ void __launch_bounds__(256)
row_stats(const float* __restrict__ Q, const float* __restrict__ K,
          float* __restrict__ q2, float* __restrict__ k2, float* __restrict__ wqm) {
  int idx = blockIdx.x * 256 + threadIdx.x;   // (b*4+h)*256 + n, 8192 total
  int n = idx & 255, bh = idx >> 8;
  int b = bh >> 2, h = bh & 3;
  const float* qr = Q + ((b * 256 + n) * 256 + h * 64);
  const float* kr = K + ((b * 256 + n) * 256 + h * 64);
  float s1 = 0.f, s2 = 0.f, sk = 0.f;
  for (int d = 0; d < 64; ++d) {
    float q = qr[d]; s1 += q; s2 += q * q;
    float kv = kr[d]; sk += kv * kv;
  }
  float m_ = s1 * (1.f / 64.f);
  float var = s2 * (1.f / 64.f) - m_ * m_;
  float sa = 0.f;
  for (int d = 0; d < 64; ++d) sa += fabsf(qr[d] - m_);
  float nf = sa / (sqrtf(var + EPSV) + EPSV);
  q2[idx] = s2;
  k2[idx] = sk;
  wqm[idx] = 1.f / (__expf(-nf) + 1.f);      // DELTA = 1
}

// ---- Kernel 4: pre-swizzle Q -> A-fragments, K -> B-fragments (QK^T), V -> B-fragments (PV) ----
// Each fragment is reused 16x in the attention kernel; pack once so every reuse is one 32B vector load.
__global__ void __launch_bounds__(256)
pack_qkv(const float* __restrict__ Q, const float* __restrict__ K,
         const float* __restrict__ V, __bf16* __restrict__ Qpk,
         __bf16* __restrict__ Kpk, __bf16* __restrict__ Vpk) {
  int tid = blockIdx.x * 256 + threadIdx.x;   // 3 * 32768 threads, 16 bf16 per thread
  int mat = tid >> 15;                        // 0=Q, 1=K, 2=V
  int r = tid & 32767;                        // frag*32 + lane
  int lane = r & 31, frag = r >> 5;           // frag in [0,1024)
  int bh = frag >> 5;
  v16bf val;
  if (mat == 0) {                             // Q A-frag: frag = bh*32 + rt*2 + kt
    int rt = (frag >> 1) & 15, kt = frag & 1;
    int b = bh >> 2, h = bh & 3;
    const float* qrow = Q + ((b * 256 + rt * 16 + (lane & 15)) * 256 + h * 64 + kt * 32);
#pragma unroll
    for (int e = 0; e < 16; ++e) val[e] = (__bf16)qrow[a_kidx(e, lane)];
    *(v16bf*)(Qpk + r * 16) = val;
  } else if (mat == 1) {                      // K B-frag: frag = bh*32 + ct*2 + kt
    int ct = (frag >> 1) & 15, kt = frag & 1;
    int b = bh >> 2, h = bh & 3;
    const float* krow = K + ((b * 256 + ct * 16 + (lane & 15)) * 256 + h * 64 + kt * 32);
#pragma unroll
    for (int e = 0; e < 16; ++e) val[e] = (__bf16)krow[b_kidx(e, lane)];
    *(v16bf*)(Kpk + r * 16) = val;
  } else {                                    // V B-frag: frag = bh*32 + nt*8 + kt
    int nt = (frag >> 3) & 3, kt = frag & 7;
    int b = bh >> 2, h = bh & 3;
    int ncol = h * 64 + nt * 16 + (lane & 15);
#pragma unroll
    for (int e = 0; e < 16; ++e)
      val[e] = (__bf16)V[(b * 256 + kt * 32 + b_kidx(e, lane)) * 256 + ncol];
    *(v16bf*)(Vpk + r * 16) = val;
  }
}

// ---- Kernel 5: attention per (b,h): QK^T (WMMA) -> fused NFA logits -> softmax -> PV (WMMA) ----
// All fragments arrive as packed 32B vector loads. Dynamic LDS: 128.5KB (CDNA5 320KB/WGP).
__global__ void __launch_bounds__(256)
attention(const __bf16* __restrict__ Qpk, const __bf16* __restrict__ Kpk,
          const __bf16* __restrict__ Vpk, const float* __restrict__ q2,
          const float* __restrict__ k2, const float* __restrict__ wqm,
          float* __restrict__ O) {
  extern __shared__ float smem[];
  const int bh = blockIdx.x;                  // b*4+h
  const int b = bh >> 2, h = bh & 3;
  const int w = threadIdx.x >> 5, lane = threadIdx.x & 31;
  float* probs = smem + w * (16 * 256);
  float* rs = smem + 8 * 16 * 256;            // [8][16] row reciprocal sums
  const float* q2v = q2 + bh * 256;
  const float* k2v = k2 + bh * 256;
  const float* wqv = wqm + bh * 256;

  for (int rti = 0; rti < 2; ++rti) {         // uniform trip count across all waves
    const int rt = w + rti * 8;
    // Q fragments: invariant over ct -> hoisted, 2 vector loads per row-tile
    v16bf qa0 = load_frag(Qpk, (bh * 16 + rt) * 2 + 0, lane);
    v16bf qa1 = load_frag(Qpk, (bh * 16 + rt) * 2 + 1, lane);
    float q2r[8], wqr[8];
#pragma unroll
    for (int r = 0; r < 8; ++r) {
      int mg = rt * 16 + cd_row(r, lane);
      q2r[r] = q2v[mg]; wqr[r] = wqv[mg];
    }
    // scores: S = QK^T/8 - omega * wq
    for (int ct = 0; ct < 16; ++ct) {
      v8f acc = {};
      acc = wmma_bf16(qa0, load_frag(Kpk, (bh * 16 + ct) * 2 + 0, lane), acc);
      acc = wmma_bf16(qa1, load_frag(Kpk, (bh * 16 + ct) * 2 + 1, lane), acc);
      int ng = ct * 16 + (lane & 15);
      float k2c = k2v[ng];
#pragma unroll
      for (int r = 0; r < 8; ++r) {
        int m = cd_row(r, lane);
        float qk = acc[r];
        float s = qk * 0.125f - (q2r[r] + k2c - 2.f * qk) * wqr[r];  // ALPHA = 1
        probs[m * 256 + ng] = s;
      }
    }
    __syncthreads();
    // softmax per row (unnormalized exp; 1/sum applied in PV epilogue)
    {
      int row = lane & 15, half = lane >> 4;
      float* pr = probs + row * 256 + half * 128;
      float mx = -3.4e38f;
      for (int c = 0; c < 128; ++c) mx = fmaxf(mx, pr[c]);
      mx = fmaxf(mx, __shfl_xor(mx, 16, 32));
      float sum = 0.f;
      for (int c = 0; c < 128; ++c) { float ev = __expf(pr[c] - mx); pr[c] = ev; sum += ev; }
      sum += __shfl_xor(sum, 16, 32);
      if (half == 0) rs[w * 16 + row] = 1.f / sum;
    }
    __syncthreads();
    // O = P @ V_h  (16x64 per rowTile, K=256); A from LDS (vectorized ds loads), B packed
    const float* pr0 = probs + (lane & 15) * 256;
    for (int nt = 0; nt < 4; ++nt) {
      v8f acc = {};
      for (int kt = 0; kt < 8; ++kt) {
        v16bf a;
#pragma unroll
        for (int e = 0; e < 16; ++e) a[e] = (__bf16)pr0[kt * 32 + a_kidx(e, lane)];
        acc = wmma_bf16(a, load_frag(Vpk, (bh * 4 + nt) * 8 + kt, lane), acc);
      }
      int ncol = h * 64 + nt * 16 + (lane & 15);
#pragma unroll
      for (int r = 0; r < 8; ++r) {
        int m = cd_row(r, lane);
        O[(b * 256 + rt * 16 + m) * 256 + ncol] = acc[r] * rs[w * 16 + m];
      }
    }
    __syncthreads();
  }
}

// ---- Kernel 6: Y = O @ Wo + bo, then LE-fusion (softmax over group token norms) ----
__global__ void __launch_bounds__(256)
out_proj_fusion(const float* __restrict__ O, const __bf16* __restrict__ Wo_pk,
                const float* __restrict__ bo, float* __restrict__ out) {
  const int b = blockIdx.x >> 4;
  const int tile = blockIdx.x & 15;
  const int t = threadIdx.x;
  const int w = t >> 5, lane = t & 31;
  __shared__ float yt[16][256];
  __shared__ float ps[16][16];
  __shared__ float lam[16];
  const float* orow = O + ((b * 256 + tile * 16 + (lane & 15)) * 256);
  v8f acc0 = {}, acc1 = {};
  for (int kt = 0; kt < 8; ++kt) {
    v16bf a;
#pragma unroll
    for (int e = 0; e < 16; ++e) a[e] = (__bf16)orow[kt * 32 + a_kidx(e, lane)];
    v16bf b0 = load_frag(Wo_pk, (w * 2 + 0) * 8 + kt, lane);
    v16bf b1 = load_frag(Wo_pk, (w * 2 + 1) * 8 + kt, lane);
    acc0 = wmma_bf16(a, b0, acc0);
    acc1 = wmma_bf16(a, b1, acc1);
  }
#pragma unroll
  for (int r = 0; r < 8; ++r) {
    int m = cd_row(r, lane);
    int n0 = w * 32 + (lane & 15), n1 = w * 32 + 16 + (lane & 15);
    yt[m][n0] = acc0[r] + bo[n0];
    yt[m][n1] = acc1[r] + bo[n1];
  }
  __syncthreads();
  const int tk = t >> 4, seg = t & 15;
  float s2 = 0.f;
  for (int d = seg * 16; d < seg * 16 + 16; ++d) { float v = yt[tk][d]; s2 += v * v; }
  ps[tk][seg] = s2;
  __syncthreads();
  if (t < 16) {
    float S = 0.f;
    for (int s = 0; s < 16; ++s) S += ps[t][s];
    ps[t][0] = sqrtf(S);                       // E (token norm)
  }
  __syncthreads();
  if (t < 4) {                                 // group t within the tile (GS=4, BETA=1)
    float e0 = ps[t * 4 + 0][0], e1 = ps[t * 4 + 1][0];
    float e2 = ps[t * 4 + 2][0], e3 = ps[t * 4 + 3][0];
    float mx = fmaxf(fmaxf(e0, e1), fmaxf(e2, e3));
    float x0 = __expf(e0 - mx), x1 = __expf(e1 - mx);
    float x2 = __expf(e2 - mx), x3 = __expf(e3 - mx);
    float inv = 1.f / (x0 + x1 + x2 + x3);
    lam[t * 4 + 0] = x0 * inv; lam[t * 4 + 1] = x1 * inv;
    lam[t * 4 + 2] = x2 * inv; lam[t * 4 + 3] = x3 * inv;
  }
  __syncthreads();
  for (int g = 0; g < 4; ++g) {
    float v = lam[g * 4 + 0] * yt[g * 4 + 0][t] + lam[g * 4 + 1] * yt[g * 4 + 1][t]
            + lam[g * 4 + 2] * yt[g * 4 + 2][t] + lam[g * 4 + 3] * yt[g * 4 + 3][t];
    out[((b * 64) + tile * 4 + g) * 256 + t] = v;
  }
}

extern "C" void kernel_launch(void* const* d_in, const int* in_sizes, int n_in,
                              void* d_out, int out_size, void* d_ws, size_t ws_size,
                              hipStream_t stream) {
  (void)in_sizes; (void)n_in; (void)out_size; (void)ws_size;
  const float* x   = (const float*)d_in[0];
  const float* Wp  = (const float*)d_in[1];
  const float* bp  = (const float*)d_in[2];
  const float* lng = (const float*)d_in[3];
  const float* lnb = (const float*)d_in[4];
  const float* Wq  = (const float*)d_in[5];
  const float* bq  = (const float*)d_in[6];
  const float* Wk  = (const float*)d_in[7];
  const float* bk  = (const float*)d_in[8];
  const float* Wv  = (const float*)d_in[9];
  const float* bv  = (const float*)d_in[10];
  const float* Wo  = (const float*)d_in[11];
  const float* bo  = (const float*)d_in[12];
  float* out = (float*)d_out;

  char* ws = (char*)d_ws;
  size_t off = 0;
  auto salloc = [&](size_t bytes) -> char* {
    char* p = ws + off; off += (bytes + 255) & ~size_t(255); return p;
  };
  float* z    = (float*)salloc(524288 * sizeof(float));   // (B,N,D)
  float* Qb   = (float*)salloc(524288 * sizeof(float));
  float* Kb   = (float*)salloc(524288 * sizeof(float));
  float* Vb   = (float*)salloc(524288 * sizeof(float));
  float* Ob   = (float*)salloc(524288 * sizeof(float));
  float* q2   = (float*)salloc(8192 * sizeof(float));
  float* k2   = (float*)salloc(8192 * sizeof(float));
  float* wqm  = (float*)salloc(8192 * sizeof(float));
  __bf16* pk  = (__bf16*)salloc(5 * 65536 * sizeof(__bf16));
  __bf16* Qpk = (__bf16*)salloc(524288 * sizeof(__bf16));
  __bf16* Kpk = (__bf16*)salloc(524288 * sizeof(__bf16));
  __bf16* Vpk = (__bf16*)salloc(524288 * sizeof(__bf16));
  __bf16* Wp_pk = pk;
  __bf16* Wq_pk = pk + 1 * 65536;
  __bf16* Wk_pk = pk + 2 * 65536;
  __bf16* Wv_pk = pk + 3 * 65536;
  __bf16* Wo_pk = pk + 4 * 65536;

  pack_weights<<<1280, 256, 0, stream>>>(Wp, Wq, Wk, Wv, Wo, pk);
  patch_embed_nfnorm<<<128, 256, 0, stream>>>(x, Wp_pk, bp, lng, lnb, z);
  qkv_gemm<<<128, 256, 0, stream>>>(z, Wq_pk, Wk_pk, Wv_pk, bq, bk, bv, Qb, Kb, Vb);
  row_stats<<<32, 256, 0, stream>>>(Qb, Kb, q2, k2, wqm);
  pack_qkv<<<384, 256, 0, stream>>>(Qb, Kb, Vb, Qpk, Kpk, Vpk);
  size_t attn_lds = (8 * 16 * 256 + 8 * 16) * sizeof(float);   // ~128.5 KB (CDNA5 320KB LDS)
  attention<<<32, 256, attn_lds, stream>>>(Qpk, Kpk, Vpk, q2, k2, wqm, Ob);
  out_proj_fusion<<<128, 256, 0, stream>>>(Ob, Wo_pk, bo, out);
}
